// ModelX_47004122088026
// MI455X (gfx1250) — matrix operand order repaired and checked by reading, not compiled, not run on hardware
//
#include <hip/hip_runtime.h>
#include <math.h>

// =====================================================================
// CDNA5 (gfx1250): all matmuls via v_wmma_f32_16x16x32_bf16.
// 256 threads = 8 wave32 per block, 128x128x32 tiles, each wave owns a
// 64x32 patch (4x2 WMMA accumulators).  fp32 global -> bf16 LDS (RNE),
// fp32 accumulate.  Pipeline: DS-store packed regs -> barrier -> issue
// next global loads -> WMMA block -> convert loaded regs -> barrier.
// =====================================================================

typedef __attribute__((ext_vector_type(16))) __bf16 v16bf;
typedef __attribute__((ext_vector_type(2)))  __bf16 v2bf;
typedef __attribute__((ext_vector_type(2)))  float  v2f;
typedef __attribute__((ext_vector_type(8)))  float  v8f;

#define BM 128
#define BN 128
#define BK 32
#define LDSS 36   // shorts per row: 72B stride -> b64 stores 8B-aligned

// f32x2 -> packed bf16x2 via fptrunc (RNE).  Backend picks the native
// packed convert if gfx1250 has one; otherwise inlines the RNE sequence.
__device__ __forceinline__ unsigned int pack_bf16x2(float a, float b) {
    v2f f; f.x = a; f.y = b;
    v2bf h = __builtin_convertvector(f, v2bf);
    union { v2bf v; unsigned int u; } cv;
    cv.v = h;
    return cv.u;
}

__device__ __forceinline__ float gelu_erf(float x) {
    return 0.5f * x * (1.0f + erff(x * 0.70710678118654752f));
}

union FragAB { v16bf v; unsigned int u[8]; };

// C[z] = act( scale * A[z] @ B[z] + bias + res[z] )
// A: [M,K] (lda).  B: BT ? [N,K](ldb) : [K,N](ldb).  C: [M,N](ldc).
// Requires K % BK == 0 and N % 4 == 0 (true for every call site).
template <int ACT, bool BT, bool BOUNDS>
__global__ __launch_bounds__(256)
void gemm_wmma(const float* __restrict__ A, int lda, long long aB,
               const float* __restrict__ Bm, int ldb, long long bB,
               const float* __restrict__ bias,
               const float* __restrict__ res, int ldres,
               float* __restrict__ C, int ldc, long long cB,
               int M, int N, int K, float scale)
{
    __shared__ unsigned short As[BM * LDSS];
    __shared__ unsigned short Bs[BN * LDSS];

    const int t    = threadIdx.x;
    const int lane = t & 31;
    const int wave = t >> 5;        // 0..7
    const int wm   = wave >> 2;     // 0..1 : 64-row band
    const int wn   = wave & 3;      // 0..3 : 32-col band
    const int lm   = lane & 15;
    const int lh   = lane >> 4;

    const int mBase = blockIdx.y * BM;
    const int nBase = blockIdx.x * BN;
    const long long z = blockIdx.z;
    A  += z * aB;
    Bm += z * bB;
    C  += z * cB;

    // per-thread staging geometry (4-float chunks)
    const int aR = t >> 3;                    // base row for A chunks
    const int aC = (t & 7) << 2;              // k-offset within tile
    int bR, bC;
    if (BT) { bR = t >> 3;  bC = (t & 7) << 2; }   // [n][k] chunks, K-contiguous
    else    { bR = t >> 5;  bC = (t & 31) << 2; }  // [k][n] chunks, N-contiguous

    float4 aReg[4], bReg[4];
    uint2  aPk[4], bPk[4];

    auto loadTiles = [&](int kt) {
#pragma unroll
        for (int i = 0; i < 4; ++i) {
            int r = aR + i * 32, gm = mBase + r, gk = kt + aC;
            float4 v = make_float4(0.f, 0.f, 0.f, 0.f);
            if (!BOUNDS || (gm < M && gk + 3 < K))
                v = *(const float4*)(A + (long long)gm * lda + gk);
            aReg[i] = v;
        }
#pragma unroll
        for (int i = 0; i < 4; ++i) {
            float4 v = make_float4(0.f, 0.f, 0.f, 0.f);
            if (BT) {
                int nn = bR + i * 32, gn = nBase + nn, gk = kt + bC;
                if (!BOUNDS || (gn < N && gk + 3 < K))
                    v = *(const float4*)(Bm + (long long)gn * ldb + gk);
            } else {
                int kk = bR + i * 8, gk = kt + kk, gn = nBase + bC;
                if (!BOUNDS || (gn + 3 < N && gk < K))
                    v = *(const float4*)(Bm + (long long)gk * ldb + gn);
            }
            bReg[i] = v;
        }
    };

    auto convertTiles = [&]() {
#pragma unroll
        for (int i = 0; i < 4; ++i) {
            aPk[i].x = pack_bf16x2(aReg[i].x, aReg[i].y);
            aPk[i].y = pack_bf16x2(aReg[i].z, aReg[i].w);
            bPk[i].x = pack_bf16x2(bReg[i].x, bReg[i].y);
            bPk[i].y = pack_bf16x2(bReg[i].z, bReg[i].w);
        }
    };

    auto storeTiles = [&]() {
#pragma unroll
        for (int i = 0; i < 4; ++i) {
            int r = aR + i * 32;
            *(uint2*)&As[r * LDSS + aC] = aPk[i];         // ds_store_b64
        }
#pragma unroll
        for (int i = 0; i < 4; ++i) {
            if (BT) {
                int nn = bR + i * 32;
                *(uint2*)&Bs[nn * LDSS + bC] = bPk[i];    // ds_store_b64
            } else {                                       // scatter 4 rows (transpose)
                int kk = bR + i * 8;
                Bs[(bC + 0) * LDSS + kk] = (unsigned short)bPk[i].x;
                Bs[(bC + 1) * LDSS + kk] = (unsigned short)(bPk[i].x >> 16);
                Bs[(bC + 2) * LDSS + kk] = (unsigned short)bPk[i].y;
                Bs[(bC + 3) * LDSS + kk] = (unsigned short)(bPk[i].y >> 16);
            }
        }
    };

    v8f acc[4][2];
#pragma unroll
    for (int i = 0; i < 4; ++i)
#pragma unroll
        for (int j = 0; j < 2; ++j)
#pragma unroll
            for (int e = 0; e < 8; ++e) acc[i][j][e] = 0.0f;

    loadTiles(0);
    convertTiles();
    for (int kt = 0; kt < K; kt += BK) {
        storeTiles();                          // DS stores only
        __syncthreads();
        const bool more = (kt + BK) < K;
        if (more) loadTiles(kt + BK);          // global loads fly under WMMA

        FragAB af[4], bfr[2];
#pragma unroll
        for (int i = 0; i < 4; ++i) {
            const unsigned short* base = &As[(wm * 64 + i * 16 + lm) * LDSS];
#pragma unroll
            for (int q = 0; q < 8; ++q) {
                int kb = ((q >> 2) << 4) + (lh << 3) + ((q & 3) << 1);
                af[i].u[q] = *(const unsigned int*)(base + kb);
            }
        }
#pragma unroll
        for (int j = 0; j < 2; ++j) {
            const unsigned short* base = &Bs[(wn * 32 + j * 16 + lm) * LDSS];
#pragma unroll
            for (int q = 0; q < 8; ++q) {
                int kb = ((q >> 2) << 4) + (lh << 3) + ((q & 3) << 1);
                bfr[j].u[q] = *(const unsigned int*)(base + kb);
            }
        }
#pragma unroll
        for (int i = 0; i < 4; ++i)
#pragma unroll
            for (int j = 0; j < 2; ++j)
                acc[i][j] = __builtin_amdgcn_wmma_f32_16x16x32_bf16(
                    false, af[i].v, false, bfr[j].v,
                    (short)0, acc[i][j], false, false);

        if (more) convertTiles();              // loads have landed by now
        __syncthreads();
    }

    // ---- epilogue ----
#pragma unroll
    for (int i = 0; i < 4; ++i) {
#pragma unroll
        for (int j = 0; j < 2; ++j) {
            int gn = nBase + wn * 32 + j * 16 + lm;
#pragma unroll
            for (int r = 0; r < 8; ++r) {
                int gm = mBase + wm * 64 + i * 16 + lh * 8 + r;
                if (!BOUNDS || (gm < M && gn < N)) {
                    float val = acc[i][j][r] * scale;
                    if (bias) val += bias[gn];
                    if (res)  val += res[(long long)gm * ldres + gn];
                    if (ACT == 1) val = gelu_erf(val);
                    C[(long long)gm * ldc + gn] = val;
                }
            }
        }
    }
}

// ---------------------------------------------------------------------
// Elementwise / reduction helpers
// ---------------------------------------------------------------------
__global__ void softmax_rows(float* __restrict__ X, int cols) {
    __shared__ float red[256];
    float* x = X + (long long)blockIdx.x * cols;
    int t = threadIdx.x;
    float m = -1e30f;
    for (int i = t; i < cols; i += 256) m = fmaxf(m, x[i]);
    red[t] = m; __syncthreads();
    for (int o = 128; o > 0; o >>= 1) { if (t < o) red[t] = fmaxf(red[t], red[t + o]); __syncthreads(); }
    m = red[0]; __syncthreads();
    float s = 0.0f;
    for (int i = t; i < cols; i += 256) { float e = __expf(x[i] - m); x[i] = e; s += e; }
    red[t] = s; __syncthreads();
    for (int o = 128; o > 0; o >>= 1) { if (t < o) red[t] += red[t + o]; __syncthreads(); }
    float inv = 1.0f / red[0];
    for (int i = t; i < cols; i += 256) x[i] *= inv;
}

__global__ void ln_rows(const float* __restrict__ X, const float* __restrict__ g,
                        const float* __restrict__ b, float* __restrict__ Y, int H) {
    __shared__ float red[256];
    const float* x = X + (long long)blockIdx.x * H;
    float* y = Y + (long long)blockIdx.x * H;
    int t = threadIdx.x;
    float s = 0.0f;
    for (int i = t; i < H; i += 256) s += x[i];
    red[t] = s; __syncthreads();
    for (int o = 128; o > 0; o >>= 1) { if (t < o) red[t] += red[t + o]; __syncthreads(); }
    float mean = red[0] / (float)H; __syncthreads();
    float vs = 0.0f;
    for (int i = t; i < H; i += 256) { float d = x[i] - mean; vs += d * d; }
    red[t] = vs; __syncthreads();
    for (int o = 128; o > 0; o >>= 1) { if (t < o) red[t] += red[t + o]; __syncthreads(); }
    float rstd = rsqrtf(red[0] / (float)H + 1e-5f);
    for (int i = t; i < H; i += 256) y[i] = (x[i] - mean) * rstd * g[i] + b[i];
}

// population == broadcast(meta): genetic = (cnt*meta + (50-cnt)*mut)/50
__global__ void genetic_mix(const float* __restrict__ meta, const float* __restrict__ mut,
                            const float* __restrict__ U, float* __restrict__ out, int n) {
    int i4 = blockIdx.x * 256 + threadIdx.x;
    if (i4 * 4 >= n) return;
    int i = i4 * 4;
    float4 cnt = make_float4(0.f, 0.f, 0.f, 0.f);
    for (int k = 0; k < 50; ++k) {
        float4 u = *(const float4*)(U + (long long)k * n + i);
        cnt.x += (u.x < 0.8f) ? 1.f : 0.f;
        cnt.y += (u.y < 0.8f) ? 1.f : 0.f;
        cnt.z += (u.z < 0.8f) ? 1.f : 0.f;
        cnt.w += (u.w < 0.8f) ? 1.f : 0.f;
    }
    float4 me = *(const float4*)(meta + i);
    float4 mu = *(const float4*)(mut + i);
    float4 o;
    o.x = (cnt.x * me.x + (50.f - cnt.x) * mu.x) * 0.02f;
    o.y = (cnt.y * me.y + (50.f - cnt.y) * mu.y) * 0.02f;
    o.z = (cnt.z * me.z + (50.f - cnt.z) * mu.z) * 0.02f;
    o.w = (cnt.w * me.w + (50.f - cnt.w) * mu.w) * 0.02f;
    *(float4*)(out + i) = o;
}

__global__ void gelu_vec(const float* __restrict__ in, float* __restrict__ out, int n) {
    int i = blockIdx.x * 256 + threadIdx.x;
    if (i < n) out[i] = gelu_erf(in[i]);
}

__global__ void concat_prior_lk(const float* __restrict__ pvec, const float* __restrict__ lk,
                                float* __restrict__ out, int S, int H) {
    int i = blockIdx.x * 256 + threadIdx.x;
    int total = S * 2 * H;
    if (i >= total) return;
    int s = i / (2 * H), j = i % (2 * H);
    out[i] = (j < H) ? pvec[j] : lk[(long long)s * H + (j - H)];
}

__global__ void concat5(const float* __restrict__ a0, const float* __restrict__ a1,
                        const float* __restrict__ a2, const float* __restrict__ a3,
                        const float* __restrict__ a4, float* __restrict__ out,
                        int S, int H) {
    int i = blockIdx.x * 256 + threadIdx.x;
    int total = S * 5 * H;
    if (i >= total) return;
    int s = i / (5 * H), j = i % (5 * H);
    int sel = j / H, c = j % H;
    const float* src = (sel == 0) ? a0 : (sel == 1) ? a1 : (sel == 2) ? a2 : (sel == 3) ? a3 : a4;
    out[i] = src[(long long)s * H + c];
}

// ---------------------------------------------------------------------
// Host-side dispatch — only the 4 combos actually used get instantiated.
// ---------------------------------------------------------------------
static inline void GEMM(hipStream_t s,
                        const float* A, int lda, long long aB,
                        const float* B, int ldb, long long bB, bool bt,
                        const float* bias,
                        const float* res, int ldres,
                        float* C, int ldc, long long cB,
                        int M, int N, int K, int batch, float scale, int act)
{
    dim3 grid((N + BN - 1) / BN, (M + BM - 1) / BM, batch);
    dim3 block(256, 1, 1);
    bool bnd = (M % BM) != 0 || (N % BN) != 0;
#define LGO(ACT, BT, BND) gemm_wmma<ACT, BT, BND><<<grid, block, 0, s>>>( \
        A, lda, aB, B, ldb, bB, bias, res, ldres, C, ldc, cB, M, N, K, scale)
    if (act == 1)      LGO(1, false, false);   // GELU call sites are aligned, BT=false
    else if (bt)       LGO(0, true,  false);   // scores: aligned
    else if (bnd)      LGO(0, false, true);    // prior GEMV, vocab projection
    else               LGO(0, false, false);
#undef LGO
}

extern "C" void kernel_launch(void* const* d_in, const int* in_sizes, int n_in,
                              void* d_out, int out_size, void* d_ws, size_t ws_size,
                              hipStream_t stream) {
    (void)in_sizes; (void)n_in; (void)out_size; (void)ws_size;
    const int S = 512, H = 2048, NH = 16, HD = 128, V = 50000;
    const float SCL = 0.08838834764831845f; // 1/sqrt(128)

    const float* x = (const float*)d_in[0];   // [512,2048]
    const float* U = (const float*)d_in[1];   // [50,512,2048]
    enum { I_WQ = 2, I_WK, I_WV, I_WO,
           I_QOW1, I_QOB1, I_QOW2, I_QOB2,
           I_MLW1, I_MLB1, I_MLW2, I_MLB2, I_MLG, I_MLBE, I_MLAW, I_MLAB,
           I_SELW1, I_SELB1, I_SELW2, I_SELB2, I_MUTW1, I_MUTB1, I_MUTW2, I_MUTB2,
           I_EVW1, I_EVB1, I_EVW2, I_EVB2, I_EVW3, I_EVB3,
           I_AOW1, I_AOB1, I_AOW2, I_AOB2, I_AOW3, I_AOB3, I_AOW4, I_AOB4,
           I_PRW1, I_PRB1, I_PRW2, I_PRB2, I_LKW1, I_LKB1, I_LKW2, I_LKB2,
           I_POW1, I_POB1, I_POW2, I_POB2,
           I_INTW1, I_INTB1, I_INTW2, I_INTB2, I_INTG, I_INTBE,
           I_OUTW, I_OUTB };
    #define P(i) ((const float*)d_in[(i)])

    // ---- workspace bump allocator (~86 MB) ----
    float* ws = (float*)d_ws;
    size_t off = 0;
    auto alloc = [&](size_t n) { float* p = ws + off; off += (n + 63) & ~(size_t)63; return p; };
    float* t0 = alloc((size_t)S * H);
    float* t1 = alloc((size_t)S * H);
    float* t2 = alloc((size_t)S * H);
    float* t3 = alloc((size_t)S * H);
    float* t4 = alloc((size_t)S * H);
    float* t5 = alloc((size_t)S * H);
    float* u0 = alloc((size_t)S * 2 * H);
    float* u1 = alloc((size_t)S * 2 * H);
    float* q1 = alloc((size_t)S * NH * 2 * HD);   // [8192,256]
    float* sc = alloc((size_t)NH * S * S);        // [16,512,512]
    float* cb = alloc((size_t)S * 5 * H);         // [512,10240]
    float* gb = alloc(H);
    float* pv = alloc(H);
    float* out = (float*)d_out;                   // [512,50000] fp32

    // ---- QuantumAttention ----
    GEMM(stream, x, H, 0, P(I_WQ), H, 0, false, nullptr, nullptr, 0, t0, H, 0, S, H, H, 1, 1.f, 0); // q
    GEMM(stream, x, H, 0, P(I_WK), H, 0, false, nullptr, nullptr, 0, t1, H, 0, S, H, H, 1, 1.f, 0); // k
    GEMM(stream, x, H, 0, P(I_WV), H, 0, false, nullptr, nullptr, 0, t2, H, 0, S, H, H, 1, 1.f, 0); // v
    GEMM(stream, t0, HD, 0, P(I_QOW1), 2 * HD, 0, false, P(I_QOB1), nullptr, 0, q1, 2 * HD, 0, S * NH, 2 * HD, HD, 1, 1.f, 1);
    GEMM(stream, q1, 2 * HD, 0, P(I_QOW2), HD, 0, false, P(I_QOB2), nullptr, 0, t3, HD, 0, S * NH, HD, 2 * HD, 1, 1.f, 0); // qh
    GEMM(stream, t1, HD, 0, P(I_QOW1), 2 * HD, 0, false, P(I_QOB1), nullptr, 0, q1, 2 * HD, 0, S * NH, 2 * HD, HD, 1, 1.f, 1);
    GEMM(stream, q1, 2 * HD, 0, P(I_QOW2), HD, 0, false, P(I_QOB2), nullptr, 0, t0, HD, 0, S * NH, HD, 2 * HD, 1, 1.f, 0); // kh
    GEMM(stream, t3, H, HD, t0, H, HD, true, nullptr, nullptr, 0, sc, S, (long long)S * S, S, S, HD, NH, SCL, 0);
    softmax_rows<<<NH * S, 256, 0, stream>>>(sc, S);
    GEMM(stream, sc, S, (long long)S * S, t2, H, HD, false, nullptr, nullptr, 0, t1, H, HD, S, HD, S, NH, 1.f, 0);
    GEMM(stream, t1, H, 0, P(I_WO), H, 0, false, nullptr, nullptr, 0, t4, H, 0, S, H, H, 1, 1.f, 0); // ao = t4

    // ---- MetaLearningLayer ----
    GEMM(stream, t4, H, 0, P(I_MLW1), 2 * H, 0, false, P(I_MLB1), nullptr, 0, u0, 2 * H, 0, S, 2 * H, H, 1, 1.f, 1);
    GEMM(stream, u0, 2 * H, 0, P(I_MLW2), H, 0, false, P(I_MLB2), nullptr, 0, t0, H, 0, S, H, 2 * H, 1, 1.f, 0);
    ln_rows<<<S, 256, 0, stream>>>(t0, P(I_MLG), P(I_MLBE), t1, H);
    GEMM(stream, t1, H, 0, P(I_MLAW), H, 0, false, P(I_MLAB), t4, H, t3, H, 0, S, H, H, 1, 1.f, 0);  // meta = t3

    // ---- GeneticOptimizationEngine (population == broadcast(meta)) ----
    GEMM(stream, t3, H, 0, P(I_MUTW1), H, 0, false, P(I_MUTB1), nullptr, 0, t0, H, 0, S, H, H, 1, 1.f, 1);
    GEMM(stream, t0, H, 0, P(I_MUTW2), H, 0, false, P(I_MUTB2), nullptr, 0, t1, H, 0, S, H, H, 1, 1.f, 0);
    genetic_mix<<<(S * H / 4 + 255) / 256, 256, 0, stream>>>(t3, t1, U, t2, S * H);  // genetic = t2

    // ---- NeuralEvolutionEngine ----
    GEMM(stream, t2, H, 0, P(I_EVW1), 2 * H, 0, false, P(I_EVB1), nullptr, 0, u0, 2 * H, 0, S, 2 * H, H, 1, 1.f, 1);
    GEMM(stream, u0, 2 * H, 0, P(I_EVW2), 2 * H, 0, false, P(I_EVB2), nullptr, 0, u1, 2 * H, 0, S, 2 * H, 2 * H, 1, 1.f, 1);
    GEMM(stream, u1, 2 * H, 0, P(I_EVW3), H, 0, false, P(I_EVB3), nullptr, 0, t0, H, 0, S, H, 2 * H, 1, 1.f, 0);
    GEMM(stream, t0, H, 0, P(I_AOW1), H / 2, 0, false, P(I_AOB1), nullptr, 0, u0, H / 2, 0, S, H / 2, H, 1, 1.f, 0);
    GEMM(stream, u0, H / 2, 0, P(I_AOW2), H / 4, 0, false, P(I_AOB2), nullptr, 0, u1, H / 4, 0, S, H / 4, H / 2, 1, 1.f, 0);
    GEMM(stream, u1, H / 4, 0, P(I_AOW3), H / 2, 0, false, P(I_AOB3), nullptr, 0, u0, H / 2, 0, S, H / 2, H / 4, 1, 1.f, 0);
    GEMM(stream, u0, H / 2, 0, P(I_AOW4), H, 0, false, P(I_AOB4), nullptr, 0, t1, H, 0, S, H, H / 2, 1, 1.f, 0); // evo = t1

    // ---- BayesianReasoningEngine: prior_network(0) is one row vector ----
    gelu_vec<<<(H + 255) / 256, 256, 0, stream>>>(P(I_PRB1), gb, H);
    GEMM(stream, gb, H, 0, P(I_PRW2), H, 0, false, P(I_PRB2), nullptr, 0, pv, H, 0, 1, H, H, 1, 1.f, 0);
    GEMM(stream, t1, H, 0, P(I_LKW1), H, 0, false, P(I_LKB1), nullptr, 0, t0, H, 0, S, H, H, 1, 1.f, 1);
    GEMM(stream, t0, H, 0, P(I_LKW2), H, 0, false, P(I_LKB2), nullptr, 0, t5, H, 0, S, H, H, 1, 1.f, 0); // lk = t5
    concat_prior_lk<<<(S * 2 * H + 255) / 256, 256, 0, stream>>>(pv, t5, u0, S, H);
    GEMM(stream, u0, 2 * H, 0, P(I_POW1), H, 0, false, P(I_POB1), nullptr, 0, t0, H, 0, S, H, 2 * H, 1, 1.f, 1);
    GEMM(stream, t0, H, 0, P(I_POW2), H, 0, false, P(I_POB2), nullptr, 0, t5, H, 0, S, H, H, 1, 1.f, 0); // post = t5

    // ---- integration + output projection ----
    concat5<<<(S * 5 * H + 255) / 256, 256, 0, stream>>>(t4, t3, t2, t1, t5, cb, S, H);
    GEMM(stream, cb, 5 * H, 0, P(I_INTW1), 2 * H, 0, false, P(I_INTB1), nullptr, 0, u0, 2 * H, 0, S, 2 * H, 5 * H, 1, 1.f, 1);
    GEMM(stream, u0, 2 * H, 0, P(I_INTW2), H, 0, false, P(I_INTB2), nullptr, 0, t0, H, 0, S, H, 2 * H, 1, 1.f, 0);
    ln_rows<<<S, 256, 0, stream>>>(t0, P(I_INTG), P(I_INTBE), t1, H);
    GEMM(stream, t1, H, 0, P(I_OUTW), V, 0, false, P(I_OUTB), nullptr, 0, out, V, 0, S, V, H, 1, 1.f, 0);
    #undef P
}
